// AtomicaDynamics_70781061038090
// MI455X (gfx1250) — compile-verified, matrix operand order checked by reading, not run on hardware
//
#include <hip/hip_runtime.h>
#include <stdint.h>

// ---------------------------------------------------------------------------
// EGNN ligand/pocket step for MI455X (gfx1250).
// Heavy per-edge MLPs run on v_wmma_f32_16x16x32_f16 (f16 in, f32 accum).
// Wave32, 256-thread workgroups (8 waves), WMMA tiles split across waves.
// ---------------------------------------------------------------------------

#define HID 64
#define KP1 160            // padded concat dim (138 -> 160 = 5 * 32)
#define KCAT 138
#define N_LIG 2048
#define N_POC 4096
#define COORDS_RANGE 3.75f // 15/4

typedef _Float16 v16h __attribute__((ext_vector_type(16)));
typedef float    v8f  __attribute__((ext_vector_type(8)));

union Frag { v16h h; uint32_t u[8]; };

__device__ __forceinline__ float fast_rcp(float x) {
#if __has_builtin(__builtin_amdgcn_rcpf)
  return __builtin_amdgcn_rcpf(x);
#else
  return 1.0f / x;
#endif
}

__device__ __forceinline__ float silu_f(float v) {
  // v * sigmoid(v); v_rcp_f32 instead of IEEE divide (huge VALU savings)
  return v * fast_rcp(1.0f + __expf(-v));
}

__device__ __forceinline__ float fast_tanh(float x) {
#if defined(__AMDGCN__) && __has_builtin(__builtin_amdgcn_tanhf)
  return __builtin_amdgcn_tanhf(x);   // gfx1250 v_tanh_f32
#else
  return tanhf(x);
#endif
}

// K-offset (within a 32-wide K chunk) for 32-bit pair v of an f16 fragment
__device__ __forceinline__ int kpair(int v, int laneHi) {
  return ((v & 4) ? 16 : 0) + (laneHi ? 8 : 0) + 2 * (v & 3);
}

// ---------------------------------------------------------------------------
// Generic small linear: y[i,n] = act(b[n] + x[i,:K1]@w + x2[i,:K2]@w[K1:]) + res
// ---------------------------------------------------------------------------
__global__ void k_linear(const float* __restrict__ x, int ldx, int offx, int K1,
                         const float* __restrict__ x2, int ldx2, int K2,
                         const float* __restrict__ w, const float* __restrict__ bias,
                         const float* __restrict__ res,
                         float* __restrict__ y, int ldy, int offy,
                         int M, int N, int act) {
  int idx = blockIdx.x * blockDim.x + threadIdx.x;
  if (idx >= M * N) return;
  int i = idx / N, n = idx - i * N;
  float s = bias ? bias[n] : 0.0f;
  for (int k = 0; k < K1; ++k) s += x[i * ldx + offx + k] * w[k * N + n];
  if (x2) for (int k = 0; k < K2; ++k) s += x2[i * ldx2 + k] * w[(K1 + k) * N + n];
  if (act) s = silu_f(s);
  if (res) s += res[i * N + n];
  y[i * ldy + offy + n] = s;
}

// clip(+-50) + layernorm over 64 features, in place
__global__ void k_cln(float* __restrict__ h, int M) {
  int i = blockIdx.x * blockDim.x + threadIdx.x;
  if (i >= M) return;
  float mean = 0.0f;
  for (int k = 0; k < HID; ++k) {
    float v = fminf(fmaxf(h[i * HID + k], -50.0f), 50.0f);
    mean += v;
  }
  mean *= (1.0f / 64.0f);
  float var = 0.0f;
  for (int k = 0; k < HID; ++k) {
    float v = fminf(fmaxf(h[i * HID + k], -50.0f), 50.0f) - mean;
    var += v * v;
  }
  var *= (1.0f / 64.0f);
  float inv = rsqrtf(var + 1e-5f);
  for (int k = 0; k < HID; ++k) {
    float v = fminf(fmaxf(h[i * HID + k], -50.0f), 50.0f);
    h[i * HID + k] = (v - mean) * inv;
  }
}

// slice columns out of a strided matrix
__global__ void k_slice(const float* __restrict__ src, int lds_, int off,
                        float* __restrict__ dst, int cols, int rows) {
  int idx = blockIdx.x * blockDim.x + threadIdx.x;
  if (idx >= rows * cols) return;
  int r = idx / cols, c = idx - r * cols;
  dst[idx] = src[r * lds_ + off + c];
}

// h_l_t = [h_emb(64), t[mol]] per ligand atom
__global__ void k_hlt(const float* __restrict__ he, const float* __restrict__ tt,
                      float* __restrict__ out) {
  int idx = blockIdx.x * blockDim.x + threadIdx.x;
  if (idx >= N_LIG * 65) return;
  int r = idx / 65, c = idx - r * 65;
  out[idx] = (c < HID) ? he[r * HID + c] : tt[r >> 6];
}

// transpose + f16-convert weight: w (K x N, f32) -> wt (N x KP, f16), zero pad
__global__ void k_cvt(const float* __restrict__ w, _Float16* __restrict__ wt,
                      int K, int N, int KP) {
  int idx = blockIdx.x * blockDim.x + threadIdx.x;
  if (idx >= N * KP) return;
  int n = idx / KP, k = idx - n * KP;
  wt[idx] = (k < K) ? (_Float16)w[k * N + n] : (_Float16)0.0f;
}

__global__ void k_xapply(float* __restrict__ x, const float* __restrict__ xa, int n) {
  int idx = blockIdx.x * blockDim.x + threadIdx.x;
  if (idx < n) x[idx] += xa[idx];
}

__global__ void k_vel(const float* __restrict__ x0, const float* __restrict__ xll,
                      const float* __restrict__ xlp, float* __restrict__ out) {
  int idx = blockIdx.x * blockDim.x + threadIdx.x;
  if (idx >= N_LIG * 3) return;
  int i = idx / 3, d = idx - i * 3;
  out[i * 19 + d] = 0.6f * (xll[idx] - x0[idx]) + 0.4f * (xlp[idx] - x0[idx]);
}

__global__ void k_hfin(const float* __restrict__ h0, const float* __restrict__ h1,
                       float* __restrict__ hf) {
  int idx = blockIdx.x * blockDim.x + threadIdx.x;
  if (idx >= N_LIG * HID) return;
  int i = idx / HID, k = idx - i * HID;
  hf[idx] = 0.6f * h0[i * 65 + k] + 0.4f * h1[i * 65 + k];
}

// ---------------------------------------------------------------------------
// WMMA edge-MLP workhorse.
// One workgroup = one query node x one 64-edge chunk.
// mode 0: GCL message MLP -> column-sum -> atomicAdd into agg[node]
// mode 1: coord MLP -> phi = y@w3 -> atomicAdd cdiff*tanh(phi)*CR into xacc
// ---------------------------------------------------------------------------
__global__ __launch_bounds__(256)
void k_edge(int mode, int cross,
            const float* __restrict__ hq,   // N_LIG x 64 current h
            const float* __restrict__ hs,   // message source h (hq or hk)
            const float* __restrict__ xq,   // N_LIG x 3 current coords
            const float* __restrict__ xk,   // kv coords (xq or x_p)
            const float* __restrict__ x0q,  // initial ligand coords
            const float* __restrict__ x0k,  // initial kv coords
            const float* __restrict__ ee,   // 8 edge attrs
            const _Float16* __restrict__ w1t, // 64 x 160 (N-major, f16)
            const _Float16* __restrict__ w2t, // 64 x 64
            const float* __restrict__ b1,
            const float* __restrict__ b2,
            const float* __restrict__ w3,   // 64 (mode 1)
            float* __restrict__ agg,        // N_LIG x 64 (mode 0)
            float* __restrict__ xacc,       // N_LIG x 3 (mode 1)
            int degTotal, int chunks) {
  __shared__ _Float16 sX[64 * KP1];
  __shared__ _Float16 sY[64 * HID];
  __shared__ float    sM[64 * HID];
  __shared__ float    scd[64 * 3];
  __shared__ float    str_[64];

  const int t = threadIdx.x;
  const int bx = blockIdx.x;
  const int node = bx / chunks;
  const int chunk = bx - node * chunks;
  const int lane = t & 31;
  const int wave = t >> 5;
  const int laneHi = lane >> 4;
  const int lmod = lane & 15;

  __builtin_prefetch(w1t, 0, 1);
  __builtin_prefetch(w2t, 0, 1);

  // ---- phase 0: per-edge geometry (rad, d0, cdiff) ----
  if (t < 64) {
    int e = t;
    int g = chunk * 64 + e;
    bool valid = g < degTotal;
    int col;
    if (cross) {
      col = (node >> 6) * 128 + g;
    } else {
      int r = node & 63;
      int j = g + (g >= r ? 1 : 0);
      col = (node & ~63) + j;
    }
    float d0 = 0.0f, rad = 0.0f, dx[3] = {0.0f, 0.0f, 0.0f};
    if (valid) {
      for (int d = 0; d < 3; ++d) {
        float df = xq[node * 3 + d] - xk[col * 3 + d];
        dx[d] = df;
        rad += df * df;
        float d00 = x0q[node * 3 + d] - x0k[col * 3 + d];
        d0 += d00 * d00;
      }
    }
    float inv = valid ? fast_rcp(sqrtf(rad + 1e-8f) + 1.0f) : 0.0f; // NORM_CONST=1
    scd[e * 3 + 0] = dx[0] * inv;
    scd[e * 3 + 1] = dx[1] * inv;
    scd[e * 3 + 2] = dx[2] * inv;
    sM[e * 2 + 0] = valid ? rad : 0.0f;  // stash rad/d0 in sM scratch
    sM[e * 2 + 1] = valid ? d0 : 0.0f;
  }
  __syncthreads();

  // ---- phase 1: build f16 concat tile [h_row | h_col | rad | d0 | ee | 0] ----
  for (int idx = t; idx < 64 * KP1; idx += 256) {
    int e = idx / KP1, k = idx - e * KP1;
    int g = chunk * 64 + e;
    bool valid = g < degTotal;
    float v = 0.0f;
    if (valid) {
      if (k < 64) {
        v = hq[node * HID + k];
      } else if (k < 128) {
        int col;
        if (cross) {
          col = (node >> 6) * 128 + g;
        } else {
          int r = node & 63;
          int j = g + (g >= r ? 1 : 0);
          col = (node & ~63) + j;
        }
        v = hs[col * HID + (k - 64)];
      } else if (k == 128) v = sM[e * 2 + 0];
      else if (k == 129) v = sM[e * 2 + 1];
      else if (k < 138) v = ee[k - 130];
    }
    sX[idx] = (_Float16)v;
  }
  __syncthreads();

  // ---- GEMM1: (64 x 160) @ (160 x 64) via v_wmma_f32_16x16x32_f16 ----
  for (int tile = wave; tile < 16; tile += 8) {
    int mt = tile >> 2, nt = tile & 3;
    v8f acc = {0.0f, 0.0f, 0.0f, 0.0f, 0.0f, 0.0f, 0.0f, 0.0f};
    for (int kc = 0; kc < 5; ++kc) {
      Frag a, b;
      int arow = (mt * 16 + lmod) * KP1 + kc * 32;
      int brow = (nt * 16 + lmod) * KP1 + kc * 32;
#pragma unroll
      for (int v = 0; v < 8; ++v) {
        int k = kpair(v, laneHi);
        a.u[v] = *(const uint32_t*)(sX + arow + k);
        b.u[v] = *(const uint32_t*)(w1t + brow + k);
      }
      acc = __builtin_amdgcn_wmma_f32_16x16x32_f16(false, a.h, false, b.h,
                                                   (short)0, acc, false, false);
    }
#pragma unroll
    for (int r = 0; r < 8; ++r) {
      int m = mt * 16 + r + (laneHi ? 8 : 0);
      int n = nt * 16 + lmod;
      sY[m * HID + n] = (_Float16)silu_f(acc[r] + b1[n]);
    }
  }
  __syncthreads();

  // ---- GEMM2: (64 x 64) @ (64 x 64) ----
  for (int tile = wave; tile < 16; tile += 8) {
    int mt = tile >> 2, nt = tile & 3;
    v8f acc = {0.0f, 0.0f, 0.0f, 0.0f, 0.0f, 0.0f, 0.0f, 0.0f};
    for (int kc = 0; kc < 2; ++kc) {
      Frag a, b;
      int arow = (mt * 16 + lmod) * HID + kc * 32;
      int brow = (nt * 16 + lmod) * HID + kc * 32;
#pragma unroll
      for (int v = 0; v < 8; ++v) {
        int k = kpair(v, laneHi);
        a.u[v] = *(const uint32_t*)(sY + arow + k);
        b.u[v] = *(const uint32_t*)(w2t + brow + k);
      }
      acc = __builtin_amdgcn_wmma_f32_16x16x32_f16(false, a.h, false, b.h,
                                                   (short)0, acc, false, false);
    }
#pragma unroll
    for (int r = 0; r < 8; ++r) {
      int m = mt * 16 + r + (laneHi ? 8 : 0);
      int n = nt * 16 + lmod;
      bool valid = (chunk * 64 + m) < degTotal;
      sM[m * HID + n] = valid ? silu_f(acc[r] + b2[n]) : 0.0f;
    }
  }
  __syncthreads();

  // ---- tail ----
  if (mode == 0) {
    if (t < HID) {
      float s = 0.0f;
      for (int m = 0; m < 64; ++m) s += sM[m * HID + t];
      atomicAdd(&agg[node * HID + t], s * 0.01f);  // / NORM_FACTOR
    }
  } else {
    if (t < 64) {
      float s = 0.0f;
      for (int k = 0; k < HID; ++k) s += sM[t * HID + k] * w3[k];
      str_[t] = fast_tanh(s) * COORDS_RANGE;
    }
    __syncthreads();
    if (t < 3) {
      float s = 0.0f;
      for (int e = 0; e < 64; ++e) s += scd[e * 3 + t] * str_[e];
      atomicAdd(&xacc[node * 3 + t], s * 0.01f);
    }
  }
}

// ---------------------------------------------------------------------------
// Host orchestration
// ---------------------------------------------------------------------------
extern "C" void kernel_launch(void* const* d_in, const int* in_sizes, int n_in,
                              void* d_out, int out_size, void* d_ws, size_t ws_size,
                              hipStream_t stream) {
  (void)in_sizes; (void)n_in; (void)out_size; (void)ws_size;
  auto F = [&](int i) { return (const float*)d_in[i]; };

  // workspace carve-out (floats)
  float* wsf = (float*)d_ws;
  size_t o = 0;
  auto alloc = [&](size_t n) { float* p = wsf + o; o += n; return p; };
  float* x_l0   = alloc(N_LIG * 3);
  float* x_p    = alloc(N_POC * 3);
  float* x_res0 = alloc(N_LIG * 3);
  float* x_res1 = alloc(N_LIG * 3);
  float* tmp    = alloc(N_POC * 42);
  float* hle    = alloc(N_LIG * HID);
  float* h_l_t  = alloc(N_LIG * 65);
  float* h_p    = alloc(N_POC * HID);
  float* hk     = alloc(N_POC * HID);
  float* hA     = alloc(N_LIG * HID);
  float* hB     = alloc(N_LIG * HID);
  float* hout0  = alloc(N_LIG * 65);
  float* hout1  = alloc(N_LIG * 65);
  float* agg    = alloc(N_LIG * HID);
  float* yn     = alloc(N_LIG * HID);
  float* xacc   = alloc(N_LIG * 3);
  float* hfin   = alloc(N_LIG * HID);
  float* t1     = alloc(N_LIG * 32);
  _Float16* w1t16 = (_Float16*)alloc(64 * KP1 / 2 + 64);
  _Float16* w2t16 = (_Float16*)alloc(64 * HID / 2 + 64);

  auto lin = [&](const float* x, int ldx, int offx, int K1,
                 const float* x2, int ldx2, int K2,
                 const float* w, const float* b, const float* res,
                 float* y, int ldy, int offy, int M, int N, int act) {
    int tot = M * N;
    k_linear<<<(tot + 255) / 256, 256, 0, stream>>>(x, ldx, offx, K1, x2, ldx2, K2,
                                                    w, b, res, y, ldy, offy, M, N, act);
  };

  // ---- split coords; encoders ----
  k_slice<<<(N_LIG * 3 + 255) / 256, 256, 0, stream>>>(F(0), 19, 0, x_l0, 3, N_LIG);
  k_slice<<<(N_POC * 3 + 255) / 256, 256, 0, stream>>>(F(1), 24, 0, x_p, 3, N_POC);
  hipMemcpyAsync(x_res0, x_l0, N_LIG * 3 * sizeof(float), hipMemcpyDeviceToDevice, stream);
  hipMemcpyAsync(x_res1, x_l0, N_LIG * 3 * sizeof(float), hipMemcpyDeviceToDevice, stream);

  // atom encoder: 16 -> 32 -> 64, clip+LN
  lin(F(0), 19, 3, 16, nullptr, 0, 0, F(7), F(8), nullptr, tmp, 32, 0, N_LIG, 32, 1);
  lin(tmp, 32, 0, 32, nullptr, 0, 0, F(9), F(10), nullptr, hle, HID, 0, N_LIG, HID, 0);
  k_cln<<<(N_LIG + 255) / 256, 256, 0, stream>>>(hle, N_LIG);
  // ctx encoder: 21 -> 42 -> 64, clip+LN
  lin(F(1), 24, 3, 21, nullptr, 0, 0, F(11), F(12), nullptr, tmp, 42, 0, N_POC, 42, 1);
  lin(tmp, 42, 0, 42, nullptr, 0, 0, F(13), F(14), nullptr, h_p, HID, 0, N_POC, HID, 0);
  k_cln<<<(N_POC + 255) / 256, 256, 0, stream>>>(h_p, N_POC);
  // h_l_t = [h_emb, t]
  k_hlt<<<(N_LIG * 65 + 255) / 256, 256, 0, stream>>>(hle, F(2), h_l_t);

  float* xres[2] = {x_res0, x_res1};
  float* houts[2] = {hout0, hout1};

  for (int pass = 0; pass < 2; ++pass) {
    const int cross = pass;
    const float* embW = F(pass ? 108 : 20);
    const float* embB = F(pass ? 109 : 21);
    const float* outW = F(pass ? 112 : 22);
    const float* outB = F(pass ? 113 : 23);
    const int blkBase = pass ? 114 : 24;
    const float* ee = F(19) + (pass ? 0 : 8);  // edge_emb[0] for lp, [1] for ll
    const float* x0k = pass ? x_p : x_l0;
    float* xcur = xres[pass];
    const float* xkv = pass ? x_p : xcur;
    const int deg = pass ? 128 : 63;
    const int chunks = pass ? 2 : 1;

    lin(h_l_t, 65, 0, 65, nullptr, 0, 0, embW, embB, nullptr, hA, HID, 0, N_LIG, HID, 0);
    if (pass)
      lin(h_p, HID, 0, HID, nullptr, 0, 0, F(110), F(111), nullptr, hk, HID, 0, N_POC, HID, 0);

    float* hcur = hA;
    float* hoth = hB;
    for (int blk = 0; blk < 4; ++blk) {
      int bb = blkBase + 21 * blk;
      for (int g = 0; g < 2; ++g) {
        int gb = bb + 8 * g;
        k_cvt<<<(64 * KP1 + 255) / 256, 256, 0, stream>>>(F(gb + 0), w1t16, KCAT, 64, KP1);
        k_cvt<<<(64 * HID + 255) / 256, 256, 0, stream>>>(F(gb + 2), w2t16, 64, 64, HID);
        hipMemsetAsync(agg, 0, N_LIG * HID * sizeof(float), stream);
        const float* hs = cross ? hk : (const float*)hcur;
        k_edge<<<N_LIG * chunks, 256, 0, stream>>>(
            0, cross, hcur, hs, xcur, xkv, x_l0, x0k, ee, w1t16, w2t16,
            F(gb + 1), F(gb + 3), nullptr, agg, nullptr, deg, chunks);
        // node MLP: h' = h + lin2(silu(lin1([h, agg])))
        lin(hcur, HID, 0, HID, agg, HID, HID, F(gb + 4), F(gb + 5), nullptr,
            yn, HID, 0, N_LIG, HID, 1);
        lin(yn, HID, 0, HID, nullptr, 0, 0, F(gb + 6), F(gb + 7), hcur,
            hoth, HID, 0, N_LIG, HID, 0);
        float* tp = hcur; hcur = hoth; hoth = tp;
      }
      // coord layer
      int cb = bb + 16;
      k_cvt<<<(64 * KP1 + 255) / 256, 256, 0, stream>>>(F(cb + 0), w1t16, KCAT, 64, KP1);
      k_cvt<<<(64 * HID + 255) / 256, 256, 0, stream>>>(F(cb + 2), w2t16, 64, 64, HID);
      hipMemsetAsync(xacc, 0, N_LIG * 3 * sizeof(float), stream);
      const float* hs = cross ? hk : (const float*)hcur;
      k_edge<<<N_LIG * chunks, 256, 0, stream>>>(
          1, cross, hcur, hs, xcur, xkv, x_l0, x0k, ee, w1t16, w2t16,
          F(cb + 1), F(cb + 3), F(cb + 4), nullptr, xacc, deg, chunks);
      k_xapply<<<(N_LIG * 3 + 255) / 256, 256, 0, stream>>>(xcur, xacc, N_LIG * 3);
    }
    lin(hcur, HID, 0, HID, nullptr, 0, 0, outW, outB, nullptr,
        houts[pass], 65, 0, N_LIG, 65, 0);
  }

  // ---- final combine + decoder ----
  float* out = (float*)d_out;
  k_vel<<<(N_LIG * 3 + 255) / 256, 256, 0, stream>>>(x_l0, x_res0, x_res1, out);
  k_hfin<<<(N_LIG * HID + 255) / 256, 256, 0, stream>>>(hout0, hout1, hfin);
  lin(hfin, HID, 0, HID, nullptr, 0, 0, F(15), F(16), nullptr, t1, 32, 0, N_LIG, 32, 1);
  lin(t1, 32, 0, 32, nullptr, 0, 0, F(17), F(18), nullptr, out, 19, 3, N_LIG, 16, 0);
  hipMemsetAsync(out + N_LIG * 19, 0, N_POC * 24 * sizeof(float), stream);
}
